// DNAModule_31774168056108
// MI455X (gfx1250) — matrix operand order repaired and checked by reading, not compiled, not run on hardware
//
#include <hip/hip_runtime.h>

// ---------------------------------------------------------------------------
// Types
// ---------------------------------------------------------------------------
typedef __bf16 bf16_t;
typedef __attribute__((ext_vector_type(8)))  __bf16 v8bf;
typedef __attribute__((ext_vector_type(16))) __bf16 v16bf;
typedef __attribute__((ext_vector_type(8)))  float  v8f;

// Global-address-space qualified fragment pointers: force global_load_b128
// (not flat_load_b128, which ticks DScnt too) in the WMMA hot loop.
typedef __attribute__((address_space(1))) const v8bf  gv8bf;
typedef __attribute__((address_space(1))) const v16bf gv16bf;

#define BATCH 131072
#define HDIM  256
#define BN_EPS 1e-5f

// ---------------------------------------------------------------------------
// Column stats (deterministic two-stage): partial sums per block, then reduce.
// ---------------------------------------------------------------------------
__global__ __launch_bounds__(256) void colstats_bf16_part(
    const bf16_t* __restrict__ A, int rowsPerBlock,
    float* __restrict__ psum, float* __restrict__ psq)
{
    int col = threadIdx.x;              // 256 columns
    int r0 = blockIdx.x * rowsPerBlock;
    float s = 0.f, q = 0.f;
    for (int r = 0; r < rowsPerBlock; ++r) {
        float v = (float)A[(size_t)(r0 + r) * HDIM + col];
        s += v; q += v * v;
    }
    psum[(size_t)blockIdx.x * HDIM + col] = s;
    psq [(size_t)blockIdx.x * HDIM + col] = q;
}

__global__ __launch_bounds__(256) void colstats_f32_part(
    const float* __restrict__ X, int rowsPerBlock,
    float* __restrict__ psum, float* __restrict__ psq)
{
    int c0 = threadIdx.x, c1 = threadIdx.x + 256;   // 512 columns
    int r0 = blockIdx.x * rowsPerBlock;
    float s0 = 0.f, q0 = 0.f, s1 = 0.f, q1 = 0.f;
    for (int r = 0; r < rowsPerBlock; ++r) {
        const float* row = X + (size_t)(r0 + r) * 512;
        float a = row[c0], b = row[c1];
        s0 += a; q0 += a * a; s1 += b; q1 += b * b;
    }
    psum[(size_t)blockIdx.x * 512 + c0] = s0;
    psq [(size_t)blockIdx.x * 512 + c0] = q0;
    psum[(size_t)blockIdx.x * 512 + c1] = s1;
    psq [(size_t)blockIdx.x * 512 + c1] = q1;
}

__global__ __launch_bounds__(256) void stats_reduce(
    const float* __restrict__ psum, const float* __restrict__ psq,
    int nblocks, int ncols, float* __restrict__ sum, float* __restrict__ sq)
{
    int col = blockIdx.x * 256 + threadIdx.x;
    if (col >= ncols) return;
    float s = 0.f, q = 0.f;
    for (int b = 0; b < nblocks; ++b) {
        s += psum[(size_t)b * ncols + col];
        q += psq [(size_t)b * ncols + col];
    }
    sum[col] = s; sq[col] = q;
}

// ---------------------------------------------------------------------------
// Fold BN into Linear: Wp[n][k] = bf16(W[k][n] * scale[k])  (N-major for WMMA B)
//                      bias2[n] = c[n] + sum_k shift[k] * W[k][n]
// Stats may come from two concatenated tensors (split at K0).
// ---------------------------------------------------------------------------
__global__ __launch_bounds__(256) void fold_kernel(
    const float* __restrict__ W, const float* __restrict__ g,
    const float* __restrict__ b, const float* __restrict__ c,
    const float* __restrict__ sumA, const float* __restrict__ sqA,
    const float* __restrict__ sumB, const float* __restrict__ sqB,
    int K0, int K, int N, float invM,
    bf16_t* __restrict__ Wp, float* __restrict__ bias2)
{
    __shared__ float red[256];
    int n = blockIdx.x;
    int tid = threadIdx.x;
    float acc = 0.f;
    for (int k = tid; k < K; k += 256) {
        float s = (k < K0) ? sumA[k] : sumB[k - K0];
        float q = (k < K0) ? sqA[k]  : sqB[k - K0];
        float mean  = s * invM;
        float var   = q * invM - mean * mean;
        float scale = g[k] * rsqrtf(var + BN_EPS);
        float shift = b[k] - mean * scale;
        float w = W[(size_t)k * N + n];
        Wp[(size_t)n * K + k] = (bf16_t)(w * scale);
        acc += shift * w;
    }
    red[tid] = acc;
    __syncthreads();
    for (int off = 128; off >= 1; off >>= 1) {
        if (tid < off) red[tid] += red[tid + off];
        __syncthreads();
    }
    if (tid == 0) bias2[n] = c[n] + red[0];
}

// ---------------------------------------------------------------------------
// bf16 WMMA GEMM: Out[M x 256] = relu?( A[M x K] @ Wp^T + bias )
// A is virtually concatenated: cols [0,K0) from A0 (width K0), rest from A1.
// Block: 256 threads = 8 waves -> 32x256 tile (wave = 16 rows x 64 cols).
// K loop is split per source tensor so the hot loop is pure pointer-increment,
// and all fragment loads are explicitly global-address-space b128 loads.
// ---------------------------------------------------------------------------
__device__ __forceinline__ void wmma_k_loop(
    const bf16_t* __restrict__ aptr, int ksteps,
    const bf16_t* __restrict__ (&bptr)[4], v8f (&acc)[4])
{
    for (int it = 0; it < ksteps; ++it) {
        // A 16x32 fragment: this code gives each lane 16 contiguous K values (two 8-element chunks), same K order as B
        v8bf a_lo = *(gv8bf*)(aptr);
        v8bf a_hi = *(gv8bf*)(aptr + 8);
        v16bf a = __builtin_shufflevector(a_lo, a_hi,
            0,1,2,3,4,5,6,7,8,9,10,11,12,13,14,15);
        aptr += 32;
        #pragma unroll
        for (int j = 0; j < 4; ++j) {
            // B-matrix 32x16: lane holds column n, 16 contiguous K values
            v16bf bfr = *(gv16bf*)(bptr[j]);
            bptr[j] += 32;
            acc[j] = __builtin_amdgcn_wmma_f32_16x16x32_bf16(
                false, a, false, bfr, (short)0, acc[j], false, false);
        }
    }
}

__global__ __launch_bounds__(256) void gemm_bf16_kernel(
    const bf16_t* __restrict__ A0, const bf16_t* __restrict__ A1,
    int K0, int K, const bf16_t* __restrict__ Wp,
    const float* __restrict__ bias, bf16_t* __restrict__ Out, int relu)
{
    const int N = HDIM;
    int lane = threadIdx.x & 31;
    int wave = threadIdx.x >> 5;
    int rowGroup = blockIdx.x * 32 + (wave >> 2) * 16;
    int ncolBase = (wave & 3) * 64;
    int lo16  = lane & 15;
    int half  = lane >> 4;
    int mrow  = rowGroup + lo16;        // A-fragment row for this lane
    int kb    = half * 16;              // A-fragment K offset
    int khalf = half * 16;              // B-fragment K offset
    const int KA1 = K - K0;

    v8f acc[4];
    #pragma unroll
    for (int j = 0; j < 4; ++j) {
        float bj = bias[ncolBase + 16 * j + lo16];
        #pragma unroll
        for (int r = 0; r < 8; ++r) acc[j][r] = bj;
    }

    const bf16_t* bptr[4];
    #pragma unroll
    for (int j = 0; j < 4; ++j)
        bptr[j] = Wp + (size_t)(ncolBase + 16 * j + lo16) * K + khalf;

    // columns [0, K0) from A0
    wmma_k_loop(A0 + (size_t)mrow * K0 + kb, K0 >> 5, bptr, acc);
    // columns [K0, K) from A1 (virtual concat); bptr continues from K0
    if (KA1 > 0)
        wmma_k_loop(A1 + (size_t)mrow * KA1 + kb, KA1 >> 5, bptr, acc);

    #pragma unroll
    for (int j = 0; j < 4; ++j) {
        int n = ncolBase + 16 * j + lo16;
        #pragma unroll
        for (int r = 0; r < 8; ++r) {
            int m = rowGroup + r + half * 8;
            float v = acc[j][r];
            if (relu) v = v > 0.f ? v : 0.f;
            Out[(size_t)m * N + n] = (bf16_t)v;
        }
    }
}

// ---------------------------------------------------------------------------
// Final 256 -> 1 matrix-vector: out[m] = dot(H[m,:], wv) + bias[0]
// ---------------------------------------------------------------------------
__global__ __launch_bounds__(256) void mv_kernel(
    const bf16_t* __restrict__ H, const bf16_t* __restrict__ wv,
    const float* __restrict__ bias, float* __restrict__ out)
{
    int lane = threadIdx.x & 31;
    int wave = threadIdx.x >> 5;
    int row  = blockIdx.x * 8 + wave;
    float acc = 0.f;
    #pragma unroll
    for (int t = 0; t < 8; ++t) {
        int k = lane + t * 32;
        acc += (float)H[(size_t)row * HDIM + k] * (float)wv[k];
    }
    for (int off = 16; off >= 1; off >>= 1)
        acc += __shfl_xor(acc, off, 32);
    if (lane == 0) out[row] = acc + bias[0];
}

// ---------------------------------------------------------------------------
// f32 -> bf16 conversion (4 elements/thread, vectorized load)
// ---------------------------------------------------------------------------
__global__ __launch_bounds__(256) void cvt_kernel(
    const float* __restrict__ x, bf16_t* __restrict__ y)
{
    size_t i = ((size_t)blockIdx.x * 256 + threadIdx.x) * 4;
    float4 v = *(const float4*)(x + i);
    y[i + 0] = (bf16_t)v.x; y[i + 1] = (bf16_t)v.y;
    y[i + 2] = (bf16_t)v.z; y[i + 3] = (bf16_t)v.w;
}

// ---------------------------------------------------------------------------
// Host orchestration
// ---------------------------------------------------------------------------
extern "C" void kernel_launch(void* const* d_in, const int* in_sizes, int n_in,
                              void* d_out, int out_size, void* d_ws, size_t ws_size,
                              hipStream_t stream)
{
    (void)in_sizes; (void)n_in; (void)out_size; (void)ws_size;
    const int B = BATCH;
    const float invB = 1.0f / (float)B;
    const float* x = (const float*)d_in[0];
    auto F = [&](int i) { return (const float*)d_in[i]; };
    // Param leaf bases: (g1,b1,W1,c1,g2,b2,W2,c2) per tuple
    const int P_IN = 1, P_A = 9, P_B = 17, P_OUT = 25;

    // ---- workspace carve (~337 MB) ----
    char* ws = (char*)d_ws;
    const size_t SLOT = (size_t)B * HDIM * sizeof(bf16_t);   // 64 MiB
    bf16_t* slot[5];
    for (int i = 0; i < 5; ++i) slot[i] = (bf16_t*)(ws + i * SLOT);
    char* p = ws + 5 * SLOT;
    auto bump = [&](size_t bytes) { char* r = p; p += (bytes + 255) & ~(size_t)255; return r; };
    bf16_t* Wp    = (bf16_t*)bump((size_t)512 * 256 * sizeof(bf16_t));
    float*  biasB = (float*) bump(256 * sizeof(float));
    float*  psum  = (float*) bump((size_t)256 * 512 * sizeof(float));
    float*  psq   = (float*) bump((size_t)256 * 512 * sizeof(float));
    float*  xsum  = (float*) bump(512 * sizeof(float));
    float*  xsq   = (float*) bump(512 * sizeof(float));
    float*  hsum  = (float*) bump(256 * sizeof(float));
    float*  hsq   = (float*) bump(256 * sizeof(float));
    float*  tsum  = (float*) bump(256 * sizeof(float));
    float*  tsq   = (float*) bump(256 * sizeof(float));
    float*  ssum[5]; float* ssq[5];
    for (int i = 0; i < 5; ++i) { ssum[i] = (float*)bump(256 * 4); ssq[i] = (float*)bump(256 * 4); }

    // tensor placement (liveness-checked): xbf spans slot0+slot1
    bf16_t* xbf = slot[0];
    bf16_t* hb  = slot[2];
    bf16_t* t_in = slot[3];
    bf16_t* o0 = slot[4], *o1 = slot[0], *o2 = slot[1], *o3 = slot[3], *o4 = slot[4];

    auto stats256 = [&](const bf16_t* A, float* sum, float* sq) {
        colstats_bf16_part<<<256, 256, 0, stream>>>(A, B / 256, psum, psq);
        stats_reduce<<<1, 256, 0, stream>>>(psum, psq, 256, 256, sum, sq);
    };
    auto fold = [&](const float* W, const float* g, const float* b, const float* c,
                    const float* sA, const float* qA, const float* sB, const float* qB,
                    int K0, int K, int N) {
        fold_kernel<<<N, 256, 0, stream>>>(W, g, b, c, sA, qA, sB, qB, K0, K, N, invB, Wp, biasB);
    };
    auto gemm = [&](const bf16_t* A0, const bf16_t* A1, int K0, int K, bf16_t* out, int relu) {
        gemm_bf16_kernel<<<B / 32, 256, 0, stream>>>(A0, A1, K0, K, Wp, biasB, out, relu);
    };
    // One submodule: BN->Lin(->H=256)->ReLU->BN->Lin(->256), optional final relu/stats
    auto submod = [&](const bf16_t* A0, const bf16_t* A1, int K0, int K,
                      const float* sA, const float* qA, const float* sB, const float* qB,
                      int pb, bf16_t* out, int reluOut, float* osum, float* osq) {
        fold(F(pb + 2), F(pb + 0), F(pb + 1), F(pb + 3), sA, qA, sB, qB, K0, K, 256);
        gemm(A0, A1, K0, K, hb, 1);
        stats256(hb, hsum, hsq);
        fold(F(pb + 6), F(pb + 4), F(pb + 5), F(pb + 7), hsum, hsq, hsum, hsq, 256, 256, 256);
        gemm(hb, hb, 256, 256, out, reluOut);
        if (osum) stats256(out, osum, osq);
    };

    // ---- pipeline ----
    colstats_f32_part<<<256, 256, 0, stream>>>(x, B / 256, psum, psq);
    stats_reduce<<<2, 256, 0, stream>>>(psum, psq, 256, 512, xsum, xsq);
    cvt_kernel<<<(B * 512 / 4) / 256, 256, 0, stream>>>(x, xbf);

    // inputs.0 = submodule(x, params_input)  [512->256->256], no trailing relu
    submod(xbf, xbf, 512, 512, xsum, xsq, xsum, xsq, P_IN, t_in, 0, tsum, tsq);
    // step 0: op_a(inputs.0)
    submod(t_in, t_in, 256, 256, tsum, tsq, tsum, tsq, P_A, o0, 1, ssum[0], ssq[0]);
    // step 1: op_a(o0)
    submod(o0, o0, 256, 256, ssum[0], ssq[0], ssum[0], ssq[0], P_A, o1, 1, ssum[1], ssq[1]);
    // step 2: op_b(concat(o0, o1))
    submod(o0, o1, 256, 512, ssum[0], ssq[0], ssum[1], ssq[1], P_B, o2, 1, ssum[2], ssq[2]);
    // step 3: op_b(concat(o1, o2))
    submod(o1, o2, 256, 512, ssum[1], ssq[1], ssum[2], ssq[2], P_B, o3, 1, ssum[3], ssq[3]);
    // step 4: op_a(o3)
    submod(o3, o3, 256, 256, ssum[3], ssq[3], ssum[3], ssq[3], P_A, o4, 1, ssum[4], ssq[4]);

    // output module: [256->256->1], no trailing relu
    fold(F(P_OUT + 2), F(P_OUT + 0), F(P_OUT + 1), F(P_OUT + 3),
         ssum[4], ssq[4], ssum[4], ssq[4], 256, 256, 256);
    gemm(o4, o4, 256, 256, hb, 1);
    stats256(hb, hsum, hsq);
    fold(F(P_OUT + 6), F(P_OUT + 4), F(P_OUT + 5), F(P_OUT + 7),
         hsum, hsq, hsum, hsq, 256, 256, 1);
    mv_kernel<<<B / 8, 256, 0, stream>>>(hb, Wp, biasB, (float*)d_out);
}
